// Attention_61065845014909
// MI455X (gfx1250) — matrix operand order repaired and checked by reading, not compiled, not run on hardware
//
#include <hip/hip_runtime.h>
#include <math.h>

// ---------------------------------------------------------------------------
// Types / helpers for CDNA5 (gfx1250) bf16 WMMA
// ---------------------------------------------------------------------------
typedef __attribute__((ext_vector_type(16))) __bf16 v16bf;
typedef __attribute__((ext_vector_type(8)))  float  v8f;

__device__ __forceinline__ unsigned short f2bf(float f) {
  unsigned int u = __float_as_uint(f);
  u += 0x7fffu + ((u >> 16) & 1u);        // round-to-nearest-even
  return (unsigned short)(u >> 16);
}
__device__ __forceinline__ float bf2f(unsigned short h) {
  return __uint_as_float(((unsigned int)h) << 16);
}
__device__ __forceinline__ v8f v8f_zero() {
  v8f z;
#pragma unroll
  for (int i = 0; i < 8; ++i) z[i] = 0.0f;
  return z;
}
// A-fragment (16x32 bf16): lane half h, elements 0..15 -> K = h*16+e.
// Two 16B loads: p[0..7], p[8..15].
__device__ __forceinline__ v16bf ld_split16(const unsigned short* p) {
  v16bf r;
  ((uint4*)&r)[0] = *(const uint4*)(p);
  ((uint4*)&r)[1] = *(const uint4*)(p + 8);
  return r;
}
// B-fragment (32x16 bf16): lane = column, elements = 16 contiguous K values.
__device__ __forceinline__ v16bf ld_contig16(const unsigned short* p) {
  v16bf r;
  ((uint4*)&r)[0] = *(const uint4*)(p);
  ((uint4*)&r)[1] = *(const uint4*)(p + 8);
  return r;
}
#define WMMA_BF16(a, b, c) \
  __builtin_amdgcn_wmma_f32_16x16x32_bf16(false, (a), false, (b), (short)0, (c), false, false)

__device__ __forceinline__ float rowmax16(float t) {
#pragma unroll
  for (int m = 1; m < 16; m <<= 1) t = fmaxf(t, __shfl_xor(t, m, 32));
  return t;
}
__device__ __forceinline__ float rowsum16(float t) {
#pragma unroll
  for (int m = 1; m < 16; m <<= 1) t += __shfl_xor(t, m, 32);
  return t;
}

// ---------------------------------------------------------------------------
// Problem constants
// ---------------------------------------------------------------------------
#define BATCH 8
#define CH    256
#define HH    64
#define WW    64
#define NTOK  4096         // H*W
#define H2    4            // heads per branch
#define DH    32           // head dim
#define NR    256          // reduced tokens (16*16)
#define NWIN  256          // windows per image
#define SCALE 0.17677669529663687f   // 32^-0.5

// cat GEMM output column layout: [0,256)=lepe_lin [256,384)=qg [384,512)=qn
// [512,640)=k_local [640,768)=v_local
#define CAT_N 768

// ---------------------------------------------------------------------------
// K0: weight prep (transpose to Bt[n][k], convert f32 -> bf16)
// ---------------------------------------------------------------------------
__global__ void prep_kernel(const float* __restrict__ W_lepe, const float* __restrict__ b_lepe,
                            const float* __restrict__ Wq1, const float* __restrict__ Wq2,
                            const float* __restrict__ Wkv2, const float* __restrict__ Wkv1,
                            const float* __restrict__ Wproj, const float* __restrict__ sr_w,
                            unsigned short* __restrict__ Wcatt, unsigned short* __restrict__ Wkv1t,
                            unsigned short* __restrict__ Wprojt, unsigned short* __restrict__ srwt,
                            float* __restrict__ bcat) {
  const int T0 = CAT_N * 256;   // Wcatt
  const int T1 = 256 * 256;     // Wkv1t
  const int T2 = 256 * 256;     // Wprojt
  const int T3 = 256 * 4096;    // srwt
  const int TOTAL = T0 + T1 + T2 + T3 + CAT_N;
  for (int i = blockIdx.x * blockDim.x + threadIdx.x; i < TOTAL; i += gridDim.x * blockDim.x) {
    if (i < T0) {
      int n = i >> 8, k = i & 255;
      float v;
      if (n < 256)      v = W_lepe[k * 256 + n];
      else if (n < 384) v = Wq1[k * 128 + (n - 256)];
      else if (n < 512) v = Wq2[k * 128 + (n - 384)];
      else              v = Wkv2[k * 256 + (n - 512)];
      Wcatt[i] = f2bf(v);
    } else if (i < T0 + T1) {
      int j = i - T0; int n = j >> 8, k = j & 255;
      Wkv1t[j] = f2bf(Wkv1[k * 256 + n]);
    } else if (i < T0 + T1 + T2) {
      int j = i - T0 - T1; int n = j >> 8, k = j & 255;
      Wprojt[j] = f2bf(Wproj[k * 256 + n]);
    } else if (i < T0 + T1 + T2 + T3) {
      int j = i - T0 - T1 - T2;
      srwt[j] = f2bf(sr_w[j]);   // sr_w[o][ci][ky][kx] is already Bt[n=o][k]
    } else {
      int j = i - T0 - T1 - T2 - T3;
      bcat[j] = (j < 256) ? b_lepe[j] : 0.0f;
    }
  }
}

// ---------------------------------------------------------------------------
// K1: tokens[b][n][c] = bf16(x[b][c][n])
// ---------------------------------------------------------------------------
__global__ void tokens_kernel(const float* __restrict__ x, unsigned short* __restrict__ tok) {
  size_t idx = (size_t)blockIdx.x * blockDim.x + threadIdx.x;   // B*N*C
  int c = idx & 255; size_t m = idx >> 8;
  int b = (int)(m >> 12); int n = (int)(m & 4095);
  tok[idx] = f2bf(x[((size_t)b * CH + c) * NTOK + n]);
}

// ---------------------------------------------------------------------------
// K2: im2col for the stride-4 4x4 conv (patches[m][k], k = ci*16+ky*4+kx)
// ---------------------------------------------------------------------------
__global__ void patches_kernel(const float* __restrict__ x, unsigned short* __restrict__ pat) {
  size_t idx = (size_t)blockIdx.x * blockDim.x + threadIdx.x;   // 2048*4096
  int k = idx & 4095; size_t m = idx >> 12;
  int b = (int)(m >> 8); int p = (int)(m & 255);
  int oy = p >> 4, ox = p & 15;
  int ci = k >> 4; int r = k & 15; int ky = r >> 2, kx = r & 3;
  pat[idx] = f2bf(x[((size_t)b * CH + ci) * NTOK + (4 * oy + ky) * WW + 4 * ox + kx]);
}

// ---------------------------------------------------------------------------
// Generic bf16 WMMA GEMM: C[M,N] = A[M,K] * Bt[N,K]^T + bias
// One wave computes a 32x64 tile (2 A-frags x 4 B-frags, 8 accumulators).
// K-loop unrolled x2 with explicit ping-pong fragment buffers: phase X
// computes while phase Y's global_load_b128s are in flight, so the compiler
// emits partial loadcnt waits and no cross-buffer register copies.
// Requires K % 64 == 0 (K = 256 or 4096 here).
// __launch_bounds__(256,1): ~190 VGPRs live; default budget spilled.
// ---------------------------------------------------------------------------
__global__ void __launch_bounds__(256, 1)
gemm_bf16_kernel(const unsigned short* __restrict__ A,
                 const unsigned short* __restrict__ Bt,
                 const float* __restrict__ bias,
                 float* __restrict__ Cf, unsigned short* __restrict__ Cb,
                 int M, int N, int K, int ldc) {
  int wave = blockIdx.x * (blockDim.x >> 5) + (threadIdx.x >> 5);
  int tilesN = N >> 6;
  int mt = wave / tilesN;
  int nt = wave - mt * tilesN;
  if (mt * 32 >= M) return;
  int lane = threadIdx.x & 31, half = lane >> 4, nl = lane & 15;
  int row0 = mt << 5, n0 = nt << 6;

  v8f acc[2][4];
#pragma unroll
  for (int mi = 0; mi < 2; ++mi)
#pragma unroll
    for (int f = 0; f < 4; ++f) acc[mi][f] = v8f_zero();

  const unsigned short* arow0 = A + (size_t)(row0 + nl) * K + half * 16;
  const unsigned short* arow1 = arow0 + (size_t)16 * K;
  const unsigned short* bp[4];
#pragma unroll
  for (int f = 0; f < 4; ++f) bp[f] = Bt + (size_t)(n0 + f * 16 + nl) * K + half * 16;

  // phase-A buffer: k = 0
  v16bf aA0 = ld_split16(arow0);
  v16bf aA1 = ld_split16(arow1);
  v16bf bA[4];
#pragma unroll
  for (int f = 0; f < 4; ++f) bA[f] = ld_contig16(bp[f]);

  int k0 = 0;
  for (; k0 < K - 64; k0 += 64) {
    // phase-B loads (k0+32) in flight while phase-A computes
    v16bf aB0 = ld_split16(arow0 + k0 + 32);
    v16bf aB1 = ld_split16(arow1 + k0 + 32);
    v16bf bB[4];
#pragma unroll
    for (int f = 0; f < 4; ++f) bB[f] = ld_contig16(bp[f] + k0 + 32);
#pragma unroll
    for (int f = 0; f < 4; ++f) {
      acc[0][f] = WMMA_BF16(aA0, bA[f], acc[0][f]);
      acc[1][f] = WMMA_BF16(aA1, bA[f], acc[1][f]);
    }
    // phase-A loads (k0+64) in flight while phase-B computes
    aA0 = ld_split16(arow0 + k0 + 64);
    aA1 = ld_split16(arow1 + k0 + 64);
#pragma unroll
    for (int f = 0; f < 4; ++f) bA[f] = ld_contig16(bp[f] + k0 + 64);
#pragma unroll
    for (int f = 0; f < 4; ++f) {
      acc[0][f] = WMMA_BF16(aB0, bB[f], acc[0][f]);
      acc[1][f] = WMMA_BF16(aB1, bB[f], acc[1][f]);
    }
  }
  // epilogue: last 64 K (phase-A buffer holds k0, load k0+32 as phase-B)
  {
    v16bf aB0 = ld_split16(arow0 + k0 + 32);
    v16bf aB1 = ld_split16(arow1 + k0 + 32);
    v16bf bB[4];
#pragma unroll
    for (int f = 0; f < 4; ++f) bB[f] = ld_contig16(bp[f] + k0 + 32);
#pragma unroll
    for (int f = 0; f < 4; ++f) {
      acc[0][f] = WMMA_BF16(aA0, bA[f], acc[0][f]);
      acc[1][f] = WMMA_BF16(aA1, bA[f], acc[1][f]);
    }
#pragma unroll
    for (int f = 0; f < 4; ++f) {
      acc[0][f] = WMMA_BF16(aB0, bB[f], acc[0][f]);
      acc[1][f] = WMMA_BF16(aB1, bB[f], acc[1][f]);
    }
  }

#pragma unroll
  for (int f = 0; f < 4; ++f) {
    int col = n0 + f * 16 + nl;
    float bv = bias ? bias[col] : 0.0f;
#pragma unroll
    for (int mi = 0; mi < 2; ++mi) {
#pragma unroll
      for (int v = 0; v < 8; ++v) {
        int row = row0 + mi * 16 + half * 8 + v;
        float val = acc[mi][f][v] + bv;
        size_t o = (size_t)row * ldc + col;
        if (Cf) Cf[o] = val; else Cb[o] = f2bf(val);
      }
    }
  }
}

// ---------------------------------------------------------------------------
// K3b: LayerNorm (over C) + exact GELU, f32 -> bf16
// ---------------------------------------------------------------------------
__global__ void ln_gelu_kernel(const float* __restrict__ fr, const float* __restrict__ g,
                               const float* __restrict__ bt, unsigned short* __restrict__ out) {
  __shared__ float red[256];
  int row = blockIdx.x, t = threadIdx.x;
  float x = fr[(size_t)row * 256 + t];
  red[t] = x; __syncthreads();
  for (int s = 128; s > 0; s >>= 1) { if (t < s) red[t] += red[t + s]; __syncthreads(); }
  float mu = red[0] * (1.0f / 256.0f);
  __syncthreads();
  float d = x - mu;
  red[t] = d * d; __syncthreads();
  for (int s = 128; s > 0; s >>= 1) { if (t < s) red[t] += red[t + s]; __syncthreads(); }
  float var = red[0] * (1.0f / 256.0f);
  float y = d * rsqrtf(var + 1e-5f) * g[t] + bt[t];
  float ge = 0.5f * y * (1.0f + erff(y * 0.70710678118654752f));
  out[(size_t)row * 256 + t] = f2bf(ge);
}

// ---------------------------------------------------------------------------
// K3d: repack kv -> kg[b][h][nr][32] and vt[b][h][32][nr] (V transposed for PV WMMA)
// ---------------------------------------------------------------------------
__global__ void repack_kv_kernel(const unsigned short* __restrict__ kv,
                                 unsigned short* __restrict__ kg, unsigned short* __restrict__ vt) {
  size_t idx = (size_t)blockIdx.x * blockDim.x + threadIdx.x;   // 2048*256
  int c = idx & 255; size_t m = idx >> 8;
  int b = (int)(m >> 8); int nr = (int)(m & 255);
  int h = (c & 127) >> 5, d = c & 31;
  unsigned short v = kv[idx];
  if (c < 128) kg[(((size_t)b * H2 + h) * NR + nr) * DH + d] = v;
  else         vt[(((size_t)b * H2 + h) * DH + d) * NR + nr] = v;
}

__global__ void zero_f32_kernel(float* __restrict__ p, int n) {
  int i = blockIdx.x * blockDim.x + threadIdx.x;
  if (i < n) p[i] = 0.0f;
}

// ---------------------------------------------------------------------------
// K4: global branch flash attention. One wave = (b, head, 16-query tile).
// Streams 256 keys in tiles of 32: 2 WMMA for S, online softmax, LDS C->A
// fragment transpose, 2 WMMA for P*V. V fragments and the next key tile are
// loaded BEFORE the softmax VALU chain so global loads overlap exp/shuffles.
// gsig == 1/256 analytically (softmax rows sum to 1) -> never computed.
// ---------------------------------------------------------------------------
__global__ void gattn_kernel(const unsigned short* __restrict__ cat,
                             const unsigned short* __restrict__ kg,
                             const unsigned short* __restrict__ vt,
                             float* __restrict__ xg) {
  __shared__ __align__(16) unsigned short ldsP[8][16 * 32];
  int wib = threadIdx.x >> 5;
  int wave = blockIdx.x * 8 + wib;
  int bh = wave >> 8;          // 0..31
  int qt = wave & 255;
  int b = bh >> 2, h = bh & 3;
  int lane = threadIdx.x & 31, half = lane >> 4, nl = lane & 15;

  // Q A-fragment: qg lives at cat cols [256+h*32, +32)
  v16bf qa;
  {
    const unsigned short* p = cat + ((size_t)b * NTOK + qt * 16 + nl) * CAT_N + 256 + h * 32 + half * 16;
    qa = ld_split16(p);
  }
  const unsigned short* kgb = kg + ((size_t)(b * H2 + h) * NR) * DH;
  const unsigned short* vtb = vt + ((size_t)(b * H2 + h) * DH) * NR;

  v8f o0 = v8f_zero(), o1 = v8f_zero();
  float m8[8], l8[8];
#pragma unroll
  for (int v = 0; v < 8; ++v) { m8[v] = -1e30f; l8[v] = 0.0f; }

  unsigned short* Pl = ldsP[wib];
  // prologue: key tile 0 fragments
  v16bf kb0 = ld_contig16(kgb + (size_t)nl * DH + half * 16);
  v16bf kb1 = ld_contig16(kgb + (size_t)(16 + nl) * DH + half * 16);
  for (int kt = 0; kt < 8; ++kt) {
    // S = Q * K^T for 32 keys (two 16-key column groups)
    v8f s0 = WMMA_BF16(qa, kb0, v8f_zero());
    v8f s1 = WMMA_BF16(qa, kb1, v8f_zero());
    // issue V loads (and next K tile) now; they overlap the softmax VALU work
    v16bf vb0 = ld_contig16(vtb + (size_t)nl * NR + kt * 32 + half * 16);         // d 0..15
    v16bf vb1 = ld_contig16(vtb + (size_t)(16 + nl) * NR + kt * 32 + half * 16);  // d 16..31
    if (kt < 7) {
      kb0 = ld_contig16(kgb + (size_t)((kt + 1) * 32 + nl) * DH + half * 16);
      kb1 = ld_contig16(kgb + (size_t)((kt + 1) * 32 + 16 + nl) * DH + half * 16);
    }
#pragma unroll
    for (int v = 0; v < 8; ++v) {
      float a0 = s0[v] * SCALE, a1 = s1[v] * SCALE;
      float rm = rowmax16(fmaxf(a0, a1));
      float mn = fmaxf(m8[v], rm);
      float al = __expf(m8[v] - mn);
      float p0 = __expf(a0 - mn), p1 = __expf(a1 - mn);
      float rs = rowsum16(p0 + p1);
      l8[v] = l8[v] * al + rs;
      m8[v] = mn;
      o0[v] *= al; o1[v] *= al;
      int r = half * 8 + v;
      Pl[r * 32 + nl] = f2bf(p0);
      Pl[r * 32 + 16 + nl] = f2bf(p1);
    }
    asm volatile("s_wait_dscnt 0" ::: "memory");
    v16bf pa = ld_split16(Pl + nl * 32 + half * 16);   // C-layout -> A-layout transpose
    o0 = WMMA_BF16(pa, vb0, o0);
    o1 = WMMA_BF16(pa, vb1, o1);
  }
#pragma unroll
  for (int v = 0; v < 8; ++v) {
    float inv = 1.0f / l8[v];
    int tok = qt * 16 + half * 8 + v;
    size_t base = ((size_t)b * NTOK + tok) * 128 + h * 32;
    xg[base + nl]      = o0[v] * inv;
    xg[base + 16 + nl] = o1[v] * inv;
  }
}

// ---------------------------------------------------------------------------
// K5: local 4x4-window attention. Block = (b, window), 4 waves = 4 heads.
// One WMMA for S (16x16x32), softmax across lanes, 2 WMMA for P*V
// (keys padded 16->32 with zeros in LDS). lsig column sums via atomics.
// ---------------------------------------------------------------------------
__global__ void lattn_kernel(const unsigned short* __restrict__ cat,
                             float* __restrict__ xl, float* __restrict__ lsig) {
  __shared__ __align__(16) unsigned short ldsP[4][16 * 32];
  __shared__ __align__(16) unsigned short ldsV[4][32 * 32];
  int h = threadIdx.x >> 5;
  int b = blockIdx.x >> 8;
  int wdw = blockIdx.x & 255;
  int wy = wdw >> 4, wx = wdw & 15;
  int lane = threadIdx.x & 31, half = lane >> 4, nl = lane & 15;
  const unsigned short* base = cat + (size_t)b * NTOK * CAT_N;

#define WTOK(i) (((wy * 4 + ((i) >> 2)) * WW) + wx * 4 + ((i) & 3))

  // Q A-fragment (rows = window positions), qn at cat cols 384+h*32
  v16bf qa = ld_split16(base + (size_t)WTOK(nl) * CAT_N + 384 + h * 32 + half * 16);
  // K B-fragment (lane = key, elements = d), k_local at cat cols 512+h*32
  v16bf kb = ld_contig16(base + (size_t)WTOK(nl) * CAT_N + 512 + h * 32 + half * 16);
  v8f s = WMMA_BF16(qa, kb, v8f_zero());

  // stage V^T [d=lane][key 0..15, zeros 16..31] into LDS (v_local at 640+h*32)
  unsigned short* Vl = ldsV[h];
#pragma unroll
  for (int j = 0; j < 16; ++j)
    Vl[lane * 32 + j] = base[(size_t)WTOK(j) * CAT_N + 640 + h * 32 + lane];
#pragma unroll
  for (int j = 16; j < 32; ++j) Vl[lane * 32 + j] = 0;

  // softmax over the 16 keys (across the 16-lane half)
  float p8[8];
#pragma unroll
  for (int v = 0; v < 8; ++v) {
    float sv = s[v] * SCALE;
    float rm = rowmax16(sv);
    float e = __expf(sv - rm);
    float rs = rowsum16(e);
    p8[v] = e / rs;
  }
  // lsig: column sums of P (sum over rows i), accumulated over heads
  float cs = 0.0f;
#pragma unroll
  for (int v = 0; v < 8; ++v) cs += p8[v];
  cs += __shfl_xor(cs, 16, 32);
  if (lane < 16) atomicAdd(&lsig[((size_t)b * NWIN + wdw) * 16 + lane], cs);

  unsigned short* Pl = ldsP[h];
#pragma unroll
  for (int v = 0; v < 8; ++v) {
    int r = half * 8 + v;
    Pl[r * 32 + nl] = f2bf(p8[v]);
    Pl[r * 32 + 16 + nl] = 0;
  }
  asm volatile("s_wait_dscnt 0" ::: "memory");
  v16bf pa  = ld_split16(Pl + nl * 32 + half * 16);
  v16bf vb0 = ld_contig16(Vl + (size_t)nl * 32 + half * 16);          // d 0..15
  v16bf vb1 = ld_contig16(Vl + (size_t)(16 + nl) * 32 + half * 16);   // d 16..31
  v8f o0 = WMMA_BF16(pa, vb0, v8f_zero());
  v8f o1 = WMMA_BF16(pa, vb1, v8f_zero());
#pragma unroll
  for (int v = 0; v < 8; ++v) {
    int i = half * 8 + v;
    size_t ob = ((size_t)b * NTOK + WTOK(i)) * 128 + h * 32;
    xl[ob + nl]      = o0[v];
    xl[ob + 16 + nl] = o1[v];
  }
#undef WTOK
}

// ---------------------------------------------------------------------------
// K6: pin = bf16( concat(xg,xl)[c] + depthwise3x3(lepe_lin)[c] )
// lepe_lin = cat cols [0,256), stride 768; SAME padding, cross-correlation.
// ---------------------------------------------------------------------------
__global__ void fuse_pin_kernel(const float* __restrict__ xg, const float* __restrict__ xl,
                                const unsigned short* __restrict__ cat,
                                const float* __restrict__ conv_w, const float* __restrict__ conv_b,
                                unsigned short* __restrict__ pin) {
  size_t idx = (size_t)blockIdx.x * blockDim.x + threadIdx.x;   // B*N*C
  int c = idx & 255; size_t m = idx >> 8;
  int b = (int)(m >> 12); int n = (int)(m & 4095);
  int y = n >> 6, x = n & 63;
  float v = (c < 128) ? xg[m * 128 + c] : xl[m * 128 + (c - 128)];
  float acc = conv_b[c];
#pragma unroll
  for (int dy = 0; dy < 3; ++dy) {
#pragma unroll
    for (int dx = 0; dx < 3; ++dx) {
      int yy = y + dy - 1, xx = x + dx - 1;
      if (yy >= 0 && yy < HH && xx >= 0 && xx < WW)
        acc += bf2f(cat[((size_t)b * NTOK + yy * WW + xx) * CAT_N + c]) * conv_w[c * 9 + dy * 3 + dx];
    }
  }
  pin[idx] = f2bf(v + acc);
}

// ---------------------------------------------------------------------------
// K8: se1/se2.  gsig == 1/256 exactly; lsig mean = raw/(4 heads * 16 queries).
// ---------------------------------------------------------------------------
__global__ void se_kernel(const float* __restrict__ lsig, const float* __restrict__ sigma,
                          const float* __restrict__ magnitude,
                          float* __restrict__ se1, float* __restrict__ se2) {
  int idx = blockIdx.x * blockDim.x + threadIdx.x;   // B*N
  int b = idx >> 12; int n = idx & 4095;
  int y = n >> 6, x = n & 63;
  int w = (y >> 2) * 16 + (x >> 2);
  int j = (y & 3) * 4 + (x & 3);
  float l = lsig[((size_t)b * NWIN + w) * 16 + j] * (1.0f / 64.0f);
  float se = (l + (1.0f / 256.0f)) * 0.5f;
  float sg = sigma[0], mg = magnitude[0];
  float cy = (float)y - 31.5f, cx = (float)x - 31.5f;
  float Z = mg * sg * (1.0f / (2.0f * (float)M_PI)) * __expf(-0.5f * sg * (cy * cy + cx * cx)) + 1.0f;
  se *= Z;
  se1[(size_t)b * NTOK + y * WW + x] = se;
  se2[(size_t)b * NTOK + x * HH + y] = se;
}

// ---------------------------------------------------------------------------
// Launch
// ---------------------------------------------------------------------------
extern "C" void kernel_launch(void* const* d_in, const int* in_sizes, int n_in,
                              void* d_out, int out_size, void* d_ws, size_t ws_size,
                              hipStream_t stream) {
  const float* x      = (const float*)d_in[0];
  const float* W_lepe = (const float*)d_in[1];
  const float* b_lepe = (const float*)d_in[2];
  const float* conv_w = (const float*)d_in[3];
  const float* conv_b = (const float*)d_in[4];
  const float* sr_w   = (const float*)d_in[5];
  const float* sr_b   = (const float*)d_in[6];
  const float* norm_g = (const float*)d_in[7];
  const float* norm_b = (const float*)d_in[8];
  const float* Wq1    = (const float*)d_in[9];
  const float* Wkv1   = (const float*)d_in[10];
  const float* Wq2    = (const float*)d_in[11];
  const float* Wkv2   = (const float*)d_in[12];
  const float* Wproj  = (const float*)d_in[13];
  const float* bproj  = (const float*)d_in[14];
  const float* sigma  = (const float*)d_in[15];
  const float* magnit = (const float*)d_in[16];

  char* ws = (char*)d_ws;
  // workspace layout (bytes, all 256-aligned); patches region reused for xg
  unsigned short* tok    = (unsigned short*)(ws + 0);            // 16,777,216  tokens bf16
  unsigned short* cat    = (unsigned short*)(ws + 16777216);     // 50,331,648  fused proj out bf16
  unsigned short* pat    = (unsigned short*)(ws + 67108864);     // 16,777,216  im2col bf16
  float*          xg     = (float*)        (ws + 67108864);     // reuse after fr GEMM
  float*          fr     = (float*)        (ws + 83886080);     //  2,097,152  conv out f32
  unsigned short* kv     = (unsigned short*)(ws + 83886080);     // reuse after LN
  unsigned short* frn    = (unsigned short*)(ws + 85983232);     //  1,048,576  normed bf16
  unsigned short* kg     = (unsigned short*)(ws + 87031808);     //    524,288
  unsigned short* vt     = (unsigned short*)(ws + 87556096);     //    524,288
  unsigned short* Wcatt  = (unsigned short*)(ws + 88080384);     //    393,216
  unsigned short* Wkv1t  = (unsigned short*)(ws + 88473600);     //    131,072
  unsigned short* Wprojt = (unsigned short*)(ws + 88604672);     //    131,072
  unsigned short* srwt   = (unsigned short*)(ws + 88735744);     //  2,097,152
  float*          bcat   = (float*)        (ws + 90832896);     //      3,072
  float*          lsig   = (float*)        (ws + 90836992);     //    131,072
  float*          xl     = (float*)        (ws + 91361280);     // 16,777,216
  unsigned short* pin    = (unsigned short*)(ws + 108138496);    // 16,777,216

  float* fused = (float*)d_out;
  float* se1   = fused + (size_t)BATCH * NTOK * CH;   // +8,388,608
  float* se2   = se1 + (size_t)BATCH * NTOK;          // +32,768

  // 0) weights -> bf16 Bt layout
  prep_kernel<<<5379, 256, 0, stream>>>(W_lepe, b_lepe, Wq1, Wq2, Wkv2, Wkv1, Wproj, sr_w,
                                        Wcatt, Wkv1t, Wprojt, srwt, bcat);
  // 1) tokens (transpose + bf16)
  tokens_kernel<<<32768, 256, 0, stream>>>(x, tok);
  // 2) im2col patches for the stride-4 conv
  patches_kernel<<<32768, 256, 0, stream>>>(x, pat);
  // 3) fused projection GEMM: [32768,256] x [256,768] -> cat (bf16)
  //    waves = (32768/32)*(768/64) = 12288 -> 1536 blocks of 8 waves
  gemm_bf16_kernel<<<1536, 256, 0, stream>>>(tok, Wcatt, bcat, nullptr, cat,
                                             BATCH * NTOK, CAT_N, CH, CAT_N);
  // 4) sr conv as GEMM: [2048,4096] x [4096,256] -> fr (f32); 256 waves
  gemm_bf16_kernel<<<32, 256, 0, stream>>>(pat, srwt, sr_b, fr, nullptr,
                                           BATCH * NR, CH, 4096, CH);
  // 5) LayerNorm + GELU -> frn (bf16)
  ln_gelu_kernel<<<BATCH * NR, 256, 0, stream>>>(fr, norm_g, norm_b, frn);
  // 6) kv GEMM: [2048,256] x [256,256] -> kv (bf16; reuses fr region); 256 waves
  gemm_bf16_kernel<<<32, 256, 0, stream>>>(frn, Wkv1t, nullptr, nullptr, kv,
                                           BATCH * NR, CH, CH, CH);
  // 7) repack into kg / transposed vt
  repack_kv_kernel<<<2048, 256, 0, stream>>>(kv, kg, vt);
  // 8) zero lsig accumulators
  zero_f32_kernel<<<512, 256, 0, stream>>>(lsig, BATCH * NWIN * 16);
  // 9) global flash attention -> xg (reuses patches region)
  gattn_kernel<<<1024, 256, 0, stream>>>(cat, kg, vt, xg);
  // 10) local window attention -> xl, lsig
  lattn_kernel<<<2048, 128, 0, stream>>>(cat, xl, lsig);
  // 11) concat + LEPE depthwise conv -> pin (bf16)
  fuse_pin_kernel<<<32768, 256, 0, stream>>>(xg, xl, cat, conv_w, conv_b, pin);
  // 12) output projection GEMM: [32768,256] x [256,256] + bproj -> d_out (f32)
  //     waves = (32768/32)*(256/64) = 4096 -> 512 blocks
  gemm_bf16_kernel<<<512, 256, 0, stream>>>(pin, Wprojt, bproj, fused, nullptr,
                                            BATCH * NTOK, CH, CH, CH);
  // 13) se1/se2 epilogue (gsig = 1/256 analytically)
  se_kernel<<<128, 256, 0, stream>>>(lsig, sigma, magnit, se1, se2);
  (void)in_sizes; (void)n_in; (void)out_size; (void)ws_size;
}
